// CodebookQuantizer_58514634440953
// MI455X (gfx1250) — matrix operand order repaired and checked by reading, not compiled, not run on hardware
//
#include <hip/hip_runtime.h>
#include <stdint.h>

// CDNA5 (gfx1250) VQ codebook quantizer.
//  Phase 1: split x and codebook into bf16 hi/lo pairs (one-time, ws).
//  Phase 2: e2[k] = ||codebook[k]||^2.
//  Phase 3: tiled WMMA GEMM (3-term bf16 split ~= fp32 dot) with
//           double-buffered async global->LDS staging + running argmin of
//           dist = e2[k] - 2*x.e, then direct gather of selected rows.

typedef __attribute__((ext_vector_type(16))) __bf16 v16bf;
typedef __attribute__((ext_vector_type(8)))  float  v8f;

#define D      512
#define NROWS  16384   // 8*2048
#define KCODES 8192
#define MTILE  128
#define NCHUNK 128
#define KD     32      // d-slice per WMMA step
#define NSLICE ((KCODES / NCHUNK) * (D / KD))   // 1024 staged slices

// ---- async global->LDS path (gfx1250), guarded by __has_builtin ----------
#if defined(__has_builtin)
# if __has_builtin(__builtin_amdgcn_global_load_async_to_lds_b128) && \
     __has_builtin(__builtin_amdgcn_s_wait_asynccnt)
#  define USE_ASYNC 1
# endif
#endif
#ifndef USE_ASYNC
# define USE_ASYNC 0
#endif

#if USE_ASYNC
// Builtin signature (from clang diagnostics): (v4i AS1*, v4i AS3*, Ii, Ii),
// both pointers NON-const, pointee 'int __attribute__((vector_size(16)))'.
typedef int v4i __attribute__((__vector_size__(16)));
typedef __attribute__((address_space(1))) v4i* as1_v4i_ptr;
typedef __attribute__((address_space(3))) v4i* as3_v4i_ptr;

__device__ __forceinline__ void async_b128(void* lds, const void* g) {
  __builtin_amdgcn_global_load_async_to_lds_b128(
      (as1_v4i_ptr)const_cast<void*>(g), (as3_v4i_ptr)(lds), 0, 0);
}
#endif

__device__ __forceinline__ unsigned short f32_bf16_rne(float f) {
  unsigned int u = __float_as_uint(f);
  unsigned int r = u + 0x7fffu + ((u >> 16) & 1u);
  return (unsigned short)(r >> 16);
}
__device__ __forceinline__ float bf16_f32(unsigned short h) {
  return __uint_as_float(((unsigned int)h) << 16);
}

// Split f32 array into bf16(hi) and bf16(residual) arrays, 4 elems/thread.
__global__ void vq_split_kernel(const float* __restrict__ src,
                                unsigned short* __restrict__ hi,
                                unsigned short* __restrict__ lo,
                                int n4) {
  int i = blockIdx.x * blockDim.x + threadIdx.x;
  if (i >= n4) return;
  float4 v = ((const float4*)src)[i];
  ushort4 h, l;
  h.x = f32_bf16_rne(v.x); l.x = f32_bf16_rne(v.x - bf16_f32(h.x));
  h.y = f32_bf16_rne(v.y); l.y = f32_bf16_rne(v.y - bf16_f32(h.y));
  h.z = f32_bf16_rne(v.z); l.z = f32_bf16_rne(v.z - bf16_f32(h.z));
  h.w = f32_bf16_rne(v.w); l.w = f32_bf16_rne(v.w - bf16_f32(h.w));
  ((ushort4*)hi)[i] = h;
  ((ushort4*)lo)[i] = l;
}

// e2[row] = sum(codebook[row][:]^2), one 128-thread block per row.
__global__ void vq_e2_kernel(const float* __restrict__ cb,
                             float* __restrict__ e2) {
  int row = blockIdx.x;
  const float* p = cb + (size_t)row * D;
  float s = 0.f;
  for (int c = threadIdx.x; c < D; c += 128) { float f = p[c]; s += f * f; }
  for (int m = 16; m; m >>= 1) s += __shfl_xor(s, m, 32);
  __shared__ float partial[4];
  if ((threadIdx.x & 31) == 0) partial[threadIdx.x >> 5] = s;
  __syncthreads();
  if (threadIdx.x == 0)
    e2[row] = partial[0] + partial[1] + partial[2] + partial[3];
}

// LDS: double-buffered staging tiles overlapped (union) with merge scratch.
struct StageBufs {
  unsigned short Ahi[2][MTILE * KD];
  unsigned short Alo[2][MTILE * KD];
  unsigned short Bhi[2][NCHUNK * KD];
  unsigned short Blo[2][NCHUNK * KD];
};
struct MergeBufs {
  float mVal[MTILE][2];
  int   mIdx[MTILE][2];
  int   codes[MTILE];
};
union __align__(16) Smem {
  StageBufs s;
  MergeBufs m;
};

// Stage one (chunk, kd) slice into buffer `buf`: A/B hi/lo, 512 uint4 each.
__device__ __forceinline__ void stage_slice(Smem& sm, int buf,
                                            const unsigned short* __restrict__ xhi,
                                            const unsigned short* __restrict__ xlo,
                                            const unsigned short* __restrict__ ehi,
                                            const unsigned short* __restrict__ elo,
                                            int tid, int rowBase, int cBase, int kd) {
  for (int u = tid; u < (MTILE * KD) / 8; u += 256) {
    int rrow = u >> 2, seg = u & 3;
    size_t aoff = (size_t)(rowBase + rrow) * D + kd + seg * 8;
    size_t boff = (size_t)(cBase  + rrow) * D + kd + seg * 8;
#if USE_ASYNC
    async_b128(&sm.s.Ahi[buf][u * 8], xhi + aoff);
    async_b128(&sm.s.Alo[buf][u * 8], xlo + aoff);
    async_b128(&sm.s.Bhi[buf][u * 8], ehi + boff);
    async_b128(&sm.s.Blo[buf][u * 8], elo + boff);
#else
    ((uint4*)sm.s.Ahi[buf])[u] = *(const uint4*)(xhi + aoff);
    ((uint4*)sm.s.Alo[buf])[u] = *(const uint4*)(xlo + aoff);
    ((uint4*)sm.s.Bhi[buf])[u] = *(const uint4*)(ehi + boff);
    ((uint4*)sm.s.Blo[buf])[u] = *(const uint4*)(elo + boff);
#endif
  }
}

// Main kernel: block of 256 threads (8 waves) owns MTILE=128 rows.
__launch_bounds__(256)
__global__ void vq_argmin_gather_kernel(const unsigned short* __restrict__ xhi,
                                        const unsigned short* __restrict__ xlo,
                                        const unsigned short* __restrict__ ehi,
                                        const unsigned short* __restrict__ elo,
                                        const float* __restrict__ e2,
                                        const float* __restrict__ codebook,
                                        float* __restrict__ out) {
  __shared__ Smem smem;

  const int tid  = threadIdx.x;
  const int lane = tid & 31;
  const int w    = tid >> 5;
  const int wm   = w & 3;   // row group: 4 x 32 rows
  const int wn   = w >> 2;  // col group: 2 x 64 entries
  const int rowBase = blockIdx.x * MTILE;

  float bestV[2][8];
  int   bestI[2][8];
#pragma unroll
  for (int mt = 0; mt < 2; ++mt)
#pragma unroll
    for (int r = 0; r < 8; ++r) { bestV[mt][r] = 3.4e38f; bestI[mt][r] = 0; }

  union Frag { uint4 q[2]; v16bf v; };

  // Prologue: stage slice 0 into buffer 0.
  stage_slice(smem, 0, xhi, xlo, ehi, elo, tid, rowBase, 0, 0);
  int buf = 0;
  int sliceIdx = 0;

  for (int chunk = 0; chunk < KCODES / NCHUNK; ++chunk) {
    const int cBase = chunk * NCHUNK;
    v8f acc[2][4];
#pragma unroll
    for (int mt = 0; mt < 2; ++mt)
#pragma unroll
      for (int nt = 0; nt < 4; ++nt) acc[mt][nt] = (v8f){0,0,0,0,0,0,0,0};

    for (int kd = 0; kd < D; kd += KD) {
#if USE_ASYNC
      __builtin_amdgcn_s_wait_asynccnt(0);   // own async loads for `buf` done
#endif
      __syncthreads();                        // everyone's staging of `buf` done
      // Issue next slice into the other buffer; overlaps with WMMA below.
      const int nextSlice = sliceIdx + 1;
      if (nextSlice < NSLICE) {
        const int nchunk = nextSlice >> 4;    // D/KD == 16 slices per chunk
        const int nkd    = (nextSlice & 15) * KD;
        stage_slice(smem, buf ^ 1, xhi, xlo, ehi, elo, tid,
                    rowBase, nchunk * NCHUNK, nkd);
      }

      const unsigned short* sAhi = smem.s.Ahi[buf];
      const unsigned short* sAlo = smem.s.Alo[buf];
      const unsigned short* sBhi = smem.s.Bhi[buf];
      const unsigned short* sBlo = smem.s.Blo[buf];

      // A fragments (16x32 bf16): lanes 0-15 take K{0..7,16..23}, 16-31 rest.
      Frag ahi[2], alo[2];
      const int arow  = lane & 15;
      const int akoff = (lane < 16) ? 0 : 8;
#pragma unroll
      for (int mt = 0; mt < 2; ++mt) {
        int base = (wm * 32 + mt * 16 + arow) * KD;
        ahi[mt].q[0] = *(const uint4*)&sAhi[base + akoff];
        ahi[mt].q[1] = *(const uint4*)&sAhi[base + 16 + akoff];
        alo[mt].q[0] = *(const uint4*)&sAlo[base + akoff];
        alo[mt].q[1] = *(const uint4*)&sAlo[base + 16 + akoff];
      }

      const int bkoff = (lane < 16) ? 0 : 8;
#pragma unroll
      for (int nt = 0; nt < 4; ++nt) {
        int ent = (wn * 4 + nt) * 16 + (lane & 15);
        Frag bhi, blo;
        bhi.q[0] = *(const uint4*)&sBhi[ent * KD + bkoff];
        bhi.q[1] = *(const uint4*)&sBhi[ent * KD + 16 + bkoff];
        blo.q[0] = *(const uint4*)&sBlo[ent * KD + bkoff];
        blo.q[1] = *(const uint4*)&sBlo[ent * KD + 16 + bkoff];
#pragma unroll
        for (int mt = 0; mt < 2; ++mt) {
          acc[mt][nt] = __builtin_amdgcn_wmma_f32_16x16x32_bf16(
              false, ahi[mt].v, false, bhi.v, (short)0, acc[mt][nt], false, false);
          acc[mt][nt] = __builtin_amdgcn_wmma_f32_16x16x32_bf16(
              false, ahi[mt].v, false, blo.v, (short)0, acc[mt][nt], false, false);
          acc[mt][nt] = __builtin_amdgcn_wmma_f32_16x16x32_bf16(
              false, alo[mt].v, false, bhi.v, (short)0, acc[mt][nt], false, false);
        }
      }
      buf ^= 1;
      ++sliceIdx;
    }

    // dist = e2[n] - 2*dot; fold into per-lane running argmin.
#pragma unroll
    for (int nt = 0; nt < 4; ++nt) {
      int n = cBase + (wn * 4 + nt) * 16 + (lane & 15);
      float e2n = e2[n];
#pragma unroll
      for (int mt = 0; mt < 2; ++mt)
#pragma unroll
        for (int r = 0; r < 8; ++r) {
          float dist = e2n - 2.0f * acc[mt][nt][r];
          if (dist < bestV[mt][r] ||
              (dist == bestV[mt][r] && n < bestI[mt][r])) {
            bestV[mt][r] = dist; bestI[mt][r] = n;
          }
        }
    }
  }

  // All compute done; merge scratch aliases staging LDS -> barrier first.
  __syncthreads();

  // Reduce across the 16 lanes sharing a row (xor masks stay within halves).
#pragma unroll
  for (int mt = 0; mt < 2; ++mt)
#pragma unroll
    for (int r = 0; r < 8; ++r) {
      float v = bestV[mt][r]; int i = bestI[mt][r];
#pragma unroll
      for (int s = 1; s < 16; s <<= 1) {
        float ov = __shfl_xor(v, s, 32);
        int   oi = __shfl_xor(i, s, 32);
        if (ov < v || (ov == v && oi < i)) { v = ov; i = oi; }
      }
      if ((lane & 15) == 0) {
        int row = wm * 32 + mt * 16 + r + ((lane >= 16) ? 8 : 0);
        smem.m.mVal[row][wn] = v; smem.m.mIdx[row][wn] = i;
      }
    }
  __syncthreads();

  if (tid < MTILE) {
    float v0 = smem.m.mVal[tid][0], v1 = smem.m.mVal[tid][1];
    int   i0 = smem.m.mIdx[tid][0], i1 = smem.m.mIdx[tid][1];
    smem.m.codes[tid] = (v1 < v0 || (v1 == v0 && i1 < i0)) ? i1 : i0;
  }
  __syncthreads();

  // Gather: out[row] = codebook[codes[row]], coalesced float4.
  const float4* cb4  = (const float4*)codebook;
  float4*       out4 = (float4*)out;
  for (int idx = tid; idx < MTILE * (D / 4); idx += 256) {
    int row = idx >> 7;          // D/4 == 128
    int c   = idx & 127;
    out4[(size_t)(rowBase + row) * (D / 4) + c] =
        cb4[(size_t)smem.m.codes[row] * (D / 4) + c];
  }
}

extern "C" void kernel_launch(void* const* d_in, const int* in_sizes, int n_in,
                              void* d_out, int out_size, void* d_ws, size_t ws_size,
                              hipStream_t stream) {
  const float* x  = (const float*)d_in[0];   // [8,2048,512] f32
  const float* cb = (const float*)d_in[1];   // [8192,512]   f32
  float* out = (float*)d_out;

  unsigned char* ws = (unsigned char*)d_ws;
  unsigned short* xhi = (unsigned short*)ws;
  unsigned short* xlo = xhi + (size_t)NROWS * D;
  unsigned short* ehi = xlo + (size_t)NROWS * D;
  unsigned short* elo = ehi + (size_t)KCODES * D;
  float*          e2  = (float*)(elo + (size_t)KCODES * D);

  int n4x = NROWS * D / 4;   // 2,097,152 float4
  int n4e = KCODES * D / 4;  // 1,048,576 float4
  vq_split_kernel<<<n4x / 256, 256, 0, stream>>>(x, xhi, xlo, n4x);
  vq_split_kernel<<<n4e / 256, 256, 0, stream>>>(cb, ehi, elo, n4e);
  vq_e2_kernel<<<KCODES, 128, 0, stream>>>(cb, e2);
  vq_argmin_gather_kernel<<<NROWS / MTILE, 256, 0, stream>>>(
      xhi, xlo, ehi, elo, e2, cb, out);
}
